// BigBird_12240656794361
// MI455X (gfx1250) — matrix operand order repaired and checked by reading, not compiled, not run on hardware
//
#include <hip/hip_runtime.h>
#include <hip/hip_bf16.h>
#include <math.h>

#define DEV __device__ __forceinline__

// ---------------- problem constants ----------------
constexpr int Bn = 2, Sn = 4096, Hn = 768, NHn = 12, Dn = 64;
constexpr int BSn = 64, Gn = 2, Rn = 3, Ln = 2;
constexpr int FFn = 3072, Cn = 3;
constexpr int NBn = Sn / BSn;          // 64
constexpr int NKn = Gn + 3 + Rn;       // 8
constexpr int SKn = NKn * BSn;         // 512 sparse key columns
constexpr int GQ  = Gn * BSn;          // 128 global query rows
constexpr float SCALEc = 0.125f;       // 1/sqrt(64)
constexpr float NEGc = -1.0e9f;

// ---------------- WMMA helpers ----------------
typedef __attribute__((ext_vector_type(16))) __bf16 v16bf;
typedef __attribute__((ext_vector_type(8)))  float  v8f;

union BF16x16 {
  v16bf v;
  unsigned short u[16];
  uint4 q[2];
};

DEV unsigned short f2bf(float f) {
  union { float f; unsigned u; } c; c.f = f;
  unsigned x = c.u + 0x7FFFu + ((c.u >> 16) & 1u);   // round-to-nearest-even
  return (unsigned short)(x >> 16);
}

DEV v8f wmma_bf16(const BF16x16& a, const BF16x16& b, v8f c) {
  return __builtin_amdgcn_wmma_f32_16x16x32_bf16(false, a.v, false, b.v,
                                                 (short)0, c, false, false);
}

DEV float wave_sum(float v) { for (int m = 16; m; m >>= 1) v += __shfl_xor(v, m, 32); return v; }
DEV float wave_max(float v) { for (int m = 16; m; m >>= 1) v = fmaxf(v, __shfl_xor(v, m, 32)); return v; }

DEV float gelu_tanh(float x) {
  float x3 = x * x * x;
  return 0.5f * x * (1.f + tanhf(0.7978845608f * (x + 0.044715f * x3)));
}

// BigBird gathered key-block index: [global(2), window(3), random(3)]
DEV int key_block(int nb, int j, const int* __restrict__ rb) {
  if (j < Gn) return j;
  if (j < Gn + 3) {
    int w = nb + (j - Gn) - 1;
    return w < 0 ? 0 : (w > NBn - 1 ? NBn - 1 : w);
  }
  return rb[nb * Rn + (j - Gn - 3)];
}

// ---------------- weight packing: W[K,N] fp32 -> [K/32][N][32] bf16 ----------------
__global__ void pack_w(const float* __restrict__ W, unsigned short* __restrict__ out,
                       int K, int N) {
  int i = blockIdx.x * blockDim.x + threadIdx.x;
  if (i >= K * N) return;
  int k = i / N, n = i - k * N;
  out[((size_t)(k >> 5) * N + n) * 32 + (k & 31)] = f2bf(W[(size_t)k * N + n]);
}

// ---------------- V transpose per head: v[b,s,h*64+d] -> vT[(b*NH+h)*64+d][s] ----
__global__ void transpose_v(const unsigned short* __restrict__ v,
                            unsigned short* __restrict__ vT) {
  int i = blockIdx.x * blockDim.x + threadIdx.x;     // over B*S*H
  int d = i & (Dn - 1);
  int rest = i >> 6;                                 // (b*S + s)*NH + h
  int h = rest % NHn;
  int bs = rest / NHn;                               // b*S + s
  int s = bs & (Sn - 1);
  int b = bs >> 12;
  vT[(((size_t)(b * NHn + h) * Dn + d)) * Sn + s] = v[i];
}

// ---------------- embedding gather + LayerNorm (fp32 + bf16 outputs) ----------------
__global__ void embed_ln(const int* __restrict__ ids, const int* __restrict__ tt,
                         const float* __restrict__ etok, const float* __restrict__ epos,
                         const float* __restrict__ etyp, const float* __restrict__ sc,
                         const float* __restrict__ bi, float* __restrict__ x,
                         unsigned short* __restrict__ xbf) {
  int tok = blockIdx.x;                 // 0 .. B*S-1
  int spos = tok % Sn;
  int id = ids[tok], t = tt[tok];
  float vals[3]; float ls = 0.f, lq = 0.f;
#pragma unroll
  for (int j = 0; j < 3; ++j) {
    int hcol = threadIdx.x + j * 256;
    float v = etok[(size_t)id * Hn + hcol] + epos[(size_t)spos * Hn + hcol]
            + etyp[(size_t)t * Hn + hcol];
    vals[j] = v; ls += v; lq += v * v;
  }
  __shared__ float sm[8], sq[8];
  float wsu = wave_sum(ls), wqu = wave_sum(lq);
  if ((threadIdx.x & 31) == 0) { sm[threadIdx.x >> 5] = wsu; sq[threadIdx.x >> 5] = wqu; }
  __syncthreads();
  float ts = 0.f, tq = 0.f;
#pragma unroll
  for (int i = 0; i < 8; ++i) { ts += sm[i]; tq += sq[i]; }
  float mean = ts / Hn;
  float var  = tq / Hn - mean * mean;
  float inv  = rsqrtf(var + 1e-12f);
  size_t base = (size_t)tok * Hn;
#pragma unroll
  for (int j = 0; j < 3; ++j) {
    int hcol = threadIdx.x + j * 256;
    float o = (vals[j] - mean) * inv * sc[hcol] + bi[hcol];
    x[base + hcol] = o;
    xbf[base + hcol] = f2bf(o);
  }
}

// ---------------- residual add + LayerNorm (fp32 + bf16 outputs) ----------------
__global__ void add_ln(const float* __restrict__ xin, const float* __restrict__ a,
                       const float* __restrict__ sc, const float* __restrict__ bi,
                       float* __restrict__ xout, unsigned short* __restrict__ xbf) {
  int tok = blockIdx.x;
  size_t base = (size_t)tok * Hn;
  float vals[3]; float ls = 0.f, lq = 0.f;
#pragma unroll
  for (int j = 0; j < 3; ++j) {
    int hcol = threadIdx.x + j * 256;
    float v = xin[base + hcol] + a[base + hcol];
    vals[j] = v; ls += v; lq += v * v;
  }
  __shared__ float sm[8], sq[8];
  float wsu = wave_sum(ls), wqu = wave_sum(lq);
  if ((threadIdx.x & 31) == 0) { sm[threadIdx.x >> 5] = wsu; sq[threadIdx.x >> 5] = wqu; }
  __syncthreads();
  float ts = 0.f, tq = 0.f;
#pragma unroll
  for (int i = 0; i < 8; ++i) { ts += sm[i]; tq += sq[i]; }
  float mean = ts / Hn;
  float var  = tq / Hn - mean * mean;
  float inv  = rsqrtf(var + 1e-12f);
#pragma unroll
  for (int j = 0; j < 3; ++j) {
    int hcol = threadIdx.x + j * 256;
    float o = (vals[j] - mean) * inv * sc[hcol] + bi[hcol];
    xout[base + hcol] = o;
    xbf[base + hcol] = f2bf(o);
  }
}

// ---------------- WMMA GEMM: C = act(Abf16[M,K] @ Wp + bias) ----------------
// Wp packed bf16 [K/32][N][32]; one wave computes a 16x64 strip (4 accumulators,
// A fragment reused 4x, 4 independent WMMAs per k-step).
__global__ void gemm_wmma(const unsigned short* __restrict__ A,
                          const unsigned short* __restrict__ Wp,
                          const float* __restrict__ bias, void* __restrict__ Cout,
                          int M, int N, int K, int act, int outBf16, int totalWaves) {
  int wid = (blockIdx.x * blockDim.x + threadIdx.x) >> 5;
  if (wid >= totalWaves) return;
  int lane = threadIdx.x & 31, half = lane >> 4, l = lane & 15;
  int tilesN4 = N >> 6;                        // groups of 4 16-col tiles
  int tm = (wid / tilesN4) << 4;
  int tn = (wid - (wid / tilesN4) * tilesN4) << 6;
  v8f ac[4] = {v8f{}, v8f{}, v8f{}, v8f{}};
  const unsigned short* arow = A + (size_t)(tm + l) * K;
  for (int k0 = 0; k0 < K; k0 += 32) {
    BF16x16 a;
    const unsigned short* ap = arow + k0 + half * 8;
    a.q[0] = *(const uint4*)(ap);
    a.q[1] = *(const uint4*)(ap + 16);
    const unsigned short* wpb = Wp + ((size_t)(k0 >> 5) * N + tn + l) * 32 + half * 8;
    BF16x16 bm[4];
#pragma unroll
    for (int j = 0; j < 4; ++j) {
      bm[j].q[0] = *(const uint4*)(wpb + j * 512);
      bm[j].q[1] = *(const uint4*)(wpb + j * 512 + 16);
    }
#pragma unroll
    for (int j = 0; j < 4; ++j) ac[j] = wmma_bf16(a, bm[j], ac[j]);
  }
#pragma unroll
  for (int j = 0; j < 4; ++j) {
    int col = tn + j * 16 + l;
    float bv = bias[col];
#pragma unroll
    for (int i = 0; i < 8; ++i) {
      float v = ac[j][i] + bv;
      if (act == 1) v = gelu_tanh(v);
      size_t idx = (size_t)(tm + i + half * 8) * N + col;
      if (outBf16) ((unsigned short*)Cout)[idx] = f2bf(v);
      else ((float*)Cout)[idx] = v;
    }
  }
}

// ---------------- sparse attention: scores = Q Kg^T * SCALE + mask ----------------
__global__ void sparse_scores(const unsigned short* __restrict__ q,
                              const unsigned short* __restrict__ k,
                              const int* __restrict__ amask, const int* __restrict__ rb,
                              float* __restrict__ sc) {
  int wid = (blockIdx.x * blockDim.x + threadIdx.x) >> 5;
  int lane = threadIdx.x & 31, half = lane >> 4, l = lane & 15;
  int tk = wid & 31;            // 32 key tiles (512 cols)
  int tq = (wid >> 5) & 3;      // 4 query tiles (64 rows)
  int nb = (wid >> 7) & 63;
  int bh = wid >> 13;           // b*NH + h
  int h = bh % NHn, b = bh / NHn;

  int qrow = nb * BSn + tq * 16 + l;
  int kc = tk * 16 + l;
  int kb = key_block(nb, kc >> 6, rb);
  int srow = kb * BSn + (kc & 63);

  v8f acc = {};
  for (int k0 = 0; k0 < Dn; k0 += 32) {
    BF16x16 a, bm;
    const unsigned short* qp = q + ((size_t)(b * Sn + qrow)) * Hn + h * Dn + k0 + half * 8;
    a.q[0] = *(const uint4*)(qp);
    a.q[1] = *(const uint4*)(qp + 16);
    const unsigned short* kp = k + ((size_t)(b * Sn + srow)) * Hn + h * Dn + k0 + half * 8;
    bm.q[0] = *(const uint4*)(kp);
    bm.q[1] = *(const uint4*)(kp + 16);
    acc = wmma_bf16(a, bm, acc);
  }
  float mbias = (1.0f - (float)amask[b * Sn + srow]) * NEGc;
  size_t rowbase = (((size_t)bh * NBn + nb) * BSn);
#pragma unroll
  for (int i = 0; i < 8; ++i) {
    int row = tq * 16 + i + half * 8;
    sc[(rowbase + row) * SKn + tk * 16 + l] = acc[i] * SCALEc + mbias;
  }
}

// ---------------- wave-per-row softmax, fp32 in place + bf16 copy ----------------
__global__ void softmax_rows(float* __restrict__ sc, unsigned short* __restrict__ ob,
                             int rows, int Lrow) {
  int row = (blockIdx.x * blockDim.x + threadIdx.x) >> 5;
  if (row >= rows) return;
  int lane = threadIdx.x & 31;
  float* p = sc + (size_t)row * Lrow;
  unsigned short* o = ob + (size_t)row * Lrow;
  float mx = -3.4e38f;
  for (int i = lane; i < Lrow; i += 32) mx = fmaxf(mx, p[i]);
  mx = wave_max(mx);
  float sum = 0.f;
  for (int i = lane; i < Lrow; i += 32) { float e = __expf(p[i] - mx); p[i] = e; sum += e; }
  sum = wave_sum(sum);
  float inv = 1.0f / sum;
  for (int i = lane; i < Lrow; i += 32) {
    float e = p[i] * inv;
    o[i] = f2bf(e);
  }
}

// ---------------- sparse attention: out(bf16) = P @ Vg ----------------
__global__ void sparse_av(const unsigned short* __restrict__ p,
                          const unsigned short* __restrict__ vT,
                          const int* __restrict__ rb, unsigned short* __restrict__ out) {
  int wid = (blockIdx.x * blockDim.x + threadIdx.x) >> 5;
  int lane = threadIdx.x & 31, half = lane >> 4, l = lane & 15;
  int tn = wid & 3;             // 4 tiles over D=64
  int tq = (wid >> 2) & 3;      // 4 query tiles
  int nb = (wid >> 4) & 63;
  int bh = wid >> 10;
  int h = bh % NHn, b = bh / NHn;

  v8f acc = {};
  size_t prow = (((size_t)bh * NBn + nb) * BSn + tq * 16 + l) * SKn;
  const unsigned short* vrow = vT + ((size_t)bh * Dn + tn * 16 + l) * Sn;
  for (int k0 = 0; k0 < SKn; k0 += 32) {
    BF16x16 a, bm;
    const unsigned short* ap = p + prow + k0 + half * 16;
    a.q[0] = *(const uint4*)(ap);
    a.q[1] = *(const uint4*)(ap + 8);
    int c0 = k0 + half * 16;                       // 16-chunk never straddles a 64-block
    int kb = key_block(nb, c0 >> 6, rb);
    const unsigned short* vp = vrow + kb * BSn + (c0 & 63);
    bm.q[0] = *(const uint4*)(vp);
    bm.q[1] = *(const uint4*)(vp + 8);
    acc = wmma_bf16(a, bm, acc);
  }
#pragma unroll
  for (int i = 0; i < 8; ++i) {
    int s = nb * BSn + tq * 16 + i + half * 8;
    out[((size_t)(b * Sn + s)) * Hn + h * Dn + tn * 16 + l] = f2bf(acc[i]);
  }
}

// ---------------- global attention: scores over all 4096 keys ----------------
__global__ void glob_scores(const unsigned short* __restrict__ q,
                            const unsigned short* __restrict__ k,
                            const int* __restrict__ amask, float* __restrict__ scg) {
  int wid = (blockIdx.x * blockDim.x + threadIdx.x) >> 5;
  int lane = threadIdx.x & 31, half = lane >> 4, l = lane & 15;
  int tk = wid & 255;           // 256 key tiles (4096 cols)
  int tq = (wid >> 8) & 7;      // 8 query tiles (128 rows)
  int bh = wid >> 11;
  int h = bh % NHn, b = bh / NHn;

  int qrow = tq * 16 + l;       // global queries = first 128 tokens
  int srow = tk * 16 + l;
  v8f acc = {};
  for (int k0 = 0; k0 < Dn; k0 += 32) {
    BF16x16 a, bm;
    const unsigned short* qp = q + ((size_t)(b * Sn + qrow)) * Hn + h * Dn + k0 + half * 8;
    a.q[0] = *(const uint4*)(qp);
    a.q[1] = *(const uint4*)(qp + 16);
    const unsigned short* kp = k + ((size_t)(b * Sn + srow)) * Hn + h * Dn + k0 + half * 8;
    bm.q[0] = *(const uint4*)(kp);
    bm.q[1] = *(const uint4*)(kp + 16);
    acc = wmma_bf16(a, bm, acc);
  }
  float mbias = (1.0f - (float)amask[b * Sn + srow]) * NEGc;
#pragma unroll
  for (int i = 0; i < 8; ++i) {
    int row = tq * 16 + i + half * 8;
    scg[((size_t)bh * GQ + row) * Sn + tk * 16 + l] = acc[i] * SCALEc + mbias;
  }
}

// ---------------- global attention: out rows 0..127 = P @ V (dense) ----------------
__global__ void glob_av(const unsigned short* __restrict__ p,
                        const unsigned short* __restrict__ vT,
                        unsigned short* __restrict__ out) {
  int wid = (blockIdx.x * blockDim.x + threadIdx.x) >> 5;
  int lane = threadIdx.x & 31, half = lane >> 4, l = lane & 15;
  int tn = wid & 3;
  int tq = (wid >> 2) & 7;
  int bh = wid >> 5;
  int h = bh % NHn, b = bh / NHn;

  v8f acc = {};
  size_t prow = ((size_t)bh * GQ + tq * 16 + l) * Sn;
  const unsigned short* vrow = vT + ((size_t)bh * Dn + tn * 16 + l) * Sn;
  for (int k0 = 0; k0 < Sn; k0 += 32) {
    BF16x16 a, bm;
    const unsigned short* ap = p + prow + k0 + half * 16;
    a.q[0] = *(const uint4*)(ap);
    a.q[1] = *(const uint4*)(ap + 8);
    const unsigned short* vp = vrow + k0 + half * 16;
    bm.q[0] = *(const uint4*)(vp);
    bm.q[1] = *(const uint4*)(vp + 8);
    acc = wmma_bf16(a, bm, acc);
  }
#pragma unroll
  for (int i = 0; i < 8; ++i) {
    int s = tq * 16 + i + half * 8;   // 0..127 -> overwrites sparse result rows
    out[((size_t)(b * Sn + s)) * Hn + h * Dn + tn * 16 + l] = f2bf(acc[i]);
  }
}

// ---------------- classifier head (tiny: M=2) ----------------
__global__ void classifier(const float* __restrict__ x, const float* __restrict__ Wc1,
                           const float* __restrict__ bc1, const float* __restrict__ Wc2,
                           const float* __restrict__ bc2, float* __restrict__ logits) {
  int b = blockIdx.x;
  __shared__ float t[Hn];
  const float* cls = x + (size_t)b * Sn * Hn;   // token 0 of batch b
  for (int j = threadIdx.x; j < Hn; j += blockDim.x) {
    float acc = bc1[j];
    for (int i = 0; i < Hn; ++i) acc += cls[i] * Wc1[(size_t)i * Hn + j];
    t[j] = tanhf(acc);
  }
  __syncthreads();
  for (int c = threadIdx.x; c < Cn; c += blockDim.x) {
    float acc = bc2[c];
    for (int j = 0; j < Hn; ++j) acc += t[j] * Wc2[(size_t)j * Cn + c];
    logits[b * Cn + c] = acc;
  }
}

// ---------------- host orchestration ----------------
extern "C" void kernel_launch(void* const* d_in, const int* in_sizes, int n_in,
                              void* d_out, int out_size, void* d_ws, size_t ws_size,
                              hipStream_t stream) {
  (void)in_sizes; (void)n_in; (void)out_size; (void)ws_size;

  const int*   ids   = (const int*)d_in[0];
  const int*   amask = (const int*)d_in[1];
  const int*   tt    = (const int*)d_in[2];
  const int*   rb    = (const int*)d_in[3];
  const float* etok  = (const float*)d_in[4];
  const float* epos  = (const float*)d_in[5];
  const float* etyp  = (const float*)d_in[6];
  const float* lnes  = (const float*)d_in[7];
  const float* lneb  = (const float*)d_in[8];
  const float* Wq = (const float*)d_in[9];
  const float* Wk = (const float*)d_in[10];
  const float* Wv = (const float*)d_in[11];
  const float* Wo = (const float*)d_in[12];
  const float* Wf1 = (const float*)d_in[13];
  const float* Wf2 = (const float*)d_in[14];
  const float* bq = (const float*)d_in[15];
  const float* bk = (const float*)d_in[16];
  const float* bv = (const float*)d_in[17];
  const float* bo = (const float*)d_in[18];
  const float* bf1 = (const float*)d_in[19];
  const float* bf2 = (const float*)d_in[20];
  const float* ln1s = (const float*)d_in[21];
  const float* ln1b = (const float*)d_in[22];
  const float* ln2s = (const float*)d_in[23];
  const float* ln2b = (const float*)d_in[24];
  const float* Wc1 = (const float*)d_in[25];
  const float* bc1 = (const float*)d_in[26];
  const float* Wc2 = (const float*)d_in[27];
  const float* bc2 = (const float*)d_in[28];

  char* base = (char*)d_ws;
  size_t off = 0;
  auto take = [&](size_t bytes) -> char* {
    char* pp = base + off;
    off += (bytes + 255) & ~(size_t)255;
    return pp;
  };

  const int M = Bn * Sn;                              // 8192
  const size_t szX    = (size_t)M * Hn * sizeof(float);
  const size_t szXh   = (size_t)M * Hn * 2;
  const size_t szMidh = (size_t)M * FFn * 2;
  const size_t szSc   = (size_t)Bn * NHn * NBn * BSn * SKn * sizeof(float);
  const size_t szSch  = (size_t)Bn * NHn * NBn * BSn * SKn * 2;
  const size_t szScg  = (size_t)Bn * NHn * GQ * Sn * sizeof(float);
  const size_t szScgh = (size_t)Bn * NHn * GQ * Sn * 2;

  float*          x0   = (float*)take(szX);
  float*          x1   = (float*)take(szX);
  unsigned short* xb0  = (unsigned short*)take(szXh);
  unsigned short* xb1  = (unsigned short*)take(szXh);
  unsigned short* qb   = (unsigned short*)take(szXh);
  unsigned short* kbuf = (unsigned short*)take(szXh);
  unsigned short* vbuf = (unsigned short*)take(szXh);
  unsigned short* vT   = (unsigned short*)take(szXh);
  unsigned short* ao   = (unsigned short*)take(szXh);
  float*          proj = (float*)take(szX);
  unsigned short* mid  = (unsigned short*)take(szMidh);
  float*          sc   = (float*)take(szSc);
  unsigned short* pbf  = (unsigned short*)take(szSch);
  float*          scg  = (float*)take(szScg);
  unsigned short* pgbf = (unsigned short*)take(szScgh);

  unsigned short *pq[Ln], *pk[Ln], *pv[Ln], *po[Ln], *pf1[Ln], *pf2[Ln];
  for (int l = 0; l < Ln; ++l) {
    pq[l]  = (unsigned short*)take((size_t)Hn * Hn * 2);
    pk[l]  = (unsigned short*)take((size_t)Hn * Hn * 2);
    pv[l]  = (unsigned short*)take((size_t)Hn * Hn * 2);
    po[l]  = (unsigned short*)take((size_t)Hn * Hn * 2);
    pf1[l] = (unsigned short*)take((size_t)Hn * FFn * 2);
    pf2[l] = (unsigned short*)take((size_t)FFn * Hn * 2);
  }

  // ---- pack weights to bf16 WMMA layout ----
  auto packW = [&](const float* W, unsigned short* out, int K, int N) {
    int total = K * N;
    pack_w<<<(total + 255) / 256, 256, 0, stream>>>(W, out, K, N);
  };
  for (int l = 0; l < Ln; ++l) {
    packW(Wq  + (size_t)l * Hn * Hn,  pq[l],  Hn, Hn);
    packW(Wk  + (size_t)l * Hn * Hn,  pk[l],  Hn, Hn);
    packW(Wv  + (size_t)l * Hn * Hn,  pv[l],  Hn, Hn);
    packW(Wo  + (size_t)l * Hn * Hn,  po[l],  Hn, Hn);
    packW(Wf1 + (size_t)l * Hn * FFn, pf1[l], Hn, FFn);
    packW(Wf2 + (size_t)l * FFn * Hn, pf2[l], FFn, Hn);
  }

  // ---- embeddings + LN ----
  embed_ln<<<M, 256, 0, stream>>>(ids, tt, etok, epos, etyp, lnes, lneb, x0, xb0);

  float* xc = x0;          float* xa = x1;
  unsigned short* xbc = xb0; unsigned short* xba = xb1;
  const int wavesH  = (M / 16) * (Hn / 64);    // 16x64 strips: 512*12 = 6144
  const int wavesFF = (M / 16) * (FFn / 64);   // 512*48 = 24576

  for (int l = 0; l < Ln; ++l) {
    // QKV projections (bf16 outputs)
    gemm_wmma<<<wavesH / 8, 256, 0, stream>>>(xbc, pq[l], bq + (size_t)l * Hn, qb,   M, Hn, Hn, 0, 1, wavesH);
    gemm_wmma<<<wavesH / 8, 256, 0, stream>>>(xbc, pk[l], bk + (size_t)l * Hn, kbuf, M, Hn, Hn, 0, 1, wavesH);
    gemm_wmma<<<wavesH / 8, 256, 0, stream>>>(xbc, pv[l], bv + (size_t)l * Hn, vbuf, M, Hn, Hn, 0, 1, wavesH);
    transpose_v<<<(M * Hn) / 256, 256, 0, stream>>>(vbuf, vT);

    // block-sparse attention
    {
      int waves = Bn * NHn * NBn * 128;                       // 196608
      sparse_scores<<<waves / 8, 256, 0, stream>>>(qb, kbuf, amask, rb, sc);
      int rows = Bn * NHn * NBn * BSn;                        // 98304
      softmax_rows<<<rows / 8, 256, 0, stream>>>(sc, pbf, rows, SKn);
      int wavesAV = Bn * NHn * NBn * 16;                      // 24576
      sparse_av<<<wavesAV / 8, 256, 0, stream>>>(pbf, vT, rb, ao);
    }
    // global (dense) attention for first G*BS rows, overwrites those rows of ao
    {
      int waves = Bn * NHn * 8 * 256;                         // 49152
      glob_scores<<<waves / 8, 256, 0, stream>>>(qb, kbuf, amask, scg);
      int rows = Bn * NHn * GQ;                               // 3072
      softmax_rows<<<rows / 8, 256, 0, stream>>>(scg, pgbf, rows, Sn);
      int wavesAV = Bn * NHn * 8 * 4;                         // 768
      glob_av<<<wavesAV / 8, 256, 0, stream>>>(pgbf, vT, ao);
    }

    // output projection + residual LN
    gemm_wmma<<<wavesH / 8, 256, 0, stream>>>(ao, po[l], bo + (size_t)l * Hn, proj, M, Hn, Hn, 0, 0, wavesH);
    add_ln<<<M, 256, 0, stream>>>(xc, proj, ln1s + (size_t)l * Hn, ln1b + (size_t)l * Hn, xa, xba);
    { float* t = xc; xc = xa; xa = t; unsigned short* u = xbc; xbc = xba; xba = u; }

    // FFN
    gemm_wmma<<<wavesFF / 8, 256, 0, stream>>>(xbc, pf1[l], bf1 + (size_t)l * FFn, mid, M, FFn, Hn, 1, 1, wavesFF);
    gemm_wmma<<<wavesH / 8, 256, 0, stream>>>(mid, pf2[l], bf2 + (size_t)l * Hn, proj, M, Hn, FFn, 0, 0, wavesH);
    add_ln<<<M, 256, 0, stream>>>(xc, proj, ln2s + (size_t)l * Hn, ln2b + (size_t)l * Hn, xa, xba);
    { float* t = xc; xc = xa; xa = t; unsigned short* u = xbc; xbc = xba; xba = u; }
  }

  classifier<<<Bn, 256, 0, stream>>>(xc, Wc1, bc1, Wc2, bc2, (float*)d_out);
}
